// ReformerEncoder_30872224923772
// MI455X (gfx1250) — compile-verified, matrix operand order checked by reading, not run on hardware
//
#include <hip/hip_runtime.h>

// ---------------------------------------------------------------------------
// Problem constants (match the reference)
// ---------------------------------------------------------------------------
#define Bc    4
#define Tc    4096
#define Dc    512
#define Hc    8
#define DHc   64
#define NHASHc 4
#define BSc   64
#define NBc   64
#define Lc    4
#define DFFc  2048
#define BHc   (Bc*Hc)        // 32
#define Cc    (NHASHc*NBc)   // 256 chunks over concatenated rounds
#define Mrows (Bc*Tc)        // 16384
#define MASKV (-5.0e4f)

#define FLAG_RELU  1
#define FLAG_SPLIT 2

#define GBM 128
#define GBN 128
#define GBK 32

typedef __attribute__((ext_vector_type(16))) __bf16 bf16x16;
typedef __attribute__((ext_vector_type(8)))  float  f32x8;

__device__ inline f32x8 zero8() { f32x8 z = {0.f,0.f,0.f,0.f,0.f,0.f,0.f,0.f}; return z; }

// WMMA f32 = bf16 x bf16 + f32  (16x16x32)
__device__ inline f32x8 wmma_bf16(bf16x16 a, bf16x16 b, f32x8 c) {
  return __builtin_amdgcn_wmma_f32_16x16x32_bf16(false, a, false, b, (short)0, c, false, false);
}

// Load one 16x32 bf16 fragment from a row-major tile with leading dim ld.
// CDNA5 16-bit operand layout: lanes 0-15 -> rows 0-15 holding K 0..7 then
// 16..23; lanes 16-31 -> same rows holding K 8..15 then 24..31.
__device__ inline bf16x16 ldfrag(const __bf16* base, int ld) {
  int lane = threadIdx.x & 31;
  const __bf16* p = base + (size_t)(lane & 15) * ld + ((lane >> 4) << 3);
  union { bf16x16 v; uint4 q[2]; } u;
  u.q[0] = *(const uint4*)(p);
  u.q[1] = *(const uint4*)(p + 16);
  return u.v;
}

// ---------------------------------------------------------------------------
// f32 -> bf16 cast (used once per call for weights + initial x)
// ---------------------------------------------------------------------------
__global__ __launch_bounds__(256)
void cast_f32_bf16_kernel(const float* __restrict__ in, __bf16* __restrict__ out, int n) {
  int i = (blockIdx.x * 256 + threadIdx.x) * 4;
  if (i >= n) return;
  float4 f = *(const float4*)(in + i);
  out[i + 0] = (__bf16)f.x;
  out[i + 1] = (__bf16)f.y;
  out[i + 2] = (__bf16)f.z;
  out[i + 3] = (__bf16)f.w;
}

// ---------------------------------------------------------------------------
// WMMA GEMM:  Y(M,N) = X(M,K) @ W(N,K)^T  (+bias) (+relu) (+residual)
// Block tile 128x128, 8 waves (4x2), wave tile 32x64 (2x4 wmma), K step 32.
// Double-buffered LDS: one barrier per K-step; next tile's global loads are
// issued before the current tile's WMMAs.
// ---------------------------------------------------------------------------
__global__ __launch_bounds__(256)
void gemm_bf16_kernel(const __bf16* __restrict__ A, const __bf16* __restrict__ W,
                      const float* __restrict__ bias, const float* __restrict__ res,
                      float* __restrict__ outF, __bf16* __restrict__ outB,
                      int N, int K, int flags)
{
  __shared__ __bf16 As[2][GBM * GBK];
  __shared__ __bf16 Bs[2][GBN * GBK];

  int tid = threadIdx.x, lane = tid & 31, w = tid >> 5;
  int wm = w >> 1, wn = w & 1;
  int m0 = blockIdx.y * GBM, n0 = blockIdx.x * GBN;

  // 128x32 bf16 tile = 512 16B chunks -> 2 chunks per thread (rows r0, r0+64)
  int r0 = tid >> 2, kc = (tid & 3) << 3;
  uint4 ra0, ra1, rb0, rb1;

  f32x8 acc[2][4];
  #pragma unroll
  for (int mt = 0; mt < 2; ++mt)
    #pragma unroll
    for (int nt = 0; nt < 4; ++nt) acc[mt][nt] = zero8();

  // prologue: load tile 0 and stage it
  ra0 = *(const uint4*)(A + (size_t)(m0 + r0)      * K + kc);
  ra1 = *(const uint4*)(A + (size_t)(m0 + r0 + 64) * K + kc);
  rb0 = *(const uint4*)(W + (size_t)(n0 + r0)      * K + kc);
  rb1 = *(const uint4*)(W + (size_t)(n0 + r0 + 64) * K + kc);
  *(uint4*)(&As[0][r0 * GBK + kc])        = ra0;
  *(uint4*)(&As[0][(r0 + 64) * GBK + kc]) = ra1;
  *(uint4*)(&Bs[0][r0 * GBK + kc])        = rb0;
  *(uint4*)(&Bs[0][(r0 + 64) * GBK + kc]) = rb1;
  __syncthreads();

  int nk = K / GBK;
  for (int kk = 0; kk < nk; ++kk) {
    int cur = kk & 1;
    if (kk + 1 < nk) {
      int k0 = (kk + 1) * GBK;
      ra0 = *(const uint4*)(A + (size_t)(m0 + r0)      * K + k0 + kc);
      ra1 = *(const uint4*)(A + (size_t)(m0 + r0 + 64) * K + k0 + kc);
      rb0 = *(const uint4*)(W + (size_t)(n0 + r0)      * K + k0 + kc);
      rb1 = *(const uint4*)(W + (size_t)(n0 + r0 + 64) * K + k0 + kc);
      if (kk + 2 < nk) {  // CDNA5 global_prefetch_b8 of tile k+2
        __builtin_prefetch(A + (size_t)(m0 + r0) * K + k0 + GBK + kc, 0, 1);
      }
    }

    bf16x16 fa[2], fb[4];
    fa[0] = ldfrag(&As[cur][(wm * 32 +  0) * GBK], GBK);
    fa[1] = ldfrag(&As[cur][(wm * 32 + 16) * GBK], GBK);
    #pragma unroll
    for (int j = 0; j < 4; ++j)
      fb[j] = ldfrag(&Bs[cur][(wn * 64 + j * 16) * GBK], GBK);
    #pragma unroll
    for (int mt = 0; mt < 2; ++mt)
      #pragma unroll
      for (int nt = 0; nt < 4; ++nt)
        acc[mt][nt] = wmma_bf16(fa[mt], fb[nt], acc[mt][nt]);

    if (kk + 1 < nk) {
      int nxt = cur ^ 1;
      *(uint4*)(&As[nxt][r0 * GBK + kc])        = ra0;
      *(uint4*)(&As[nxt][(r0 + 64) * GBK + kc]) = ra1;
      *(uint4*)(&Bs[nxt][r0 * GBK + kc])        = rb0;
      *(uint4*)(&Bs[nxt][(r0 + 64) * GBK + kc]) = rb1;
    }
    __syncthreads();
  }

  int hi = lane >> 4, cl = lane & 15;
  #pragma unroll
  for (int mt = 0; mt < 2; ++mt) {
    #pragma unroll
    for (int nt = 0; nt < 4; ++nt) {
      #pragma unroll
      for (int i = 0; i < 8; ++i) {
        int m = m0 + wm * 32 + mt * 16 + hi * 8 + i;
        int n = n0 + wn * 64 + nt * 16 + cl;
        float v = acc[mt][nt][i];
        if (bias) v += bias[n];
        if (flags & FLAG_RELU) v = fmaxf(v, 0.f);
        if (res)  v += res[(size_t)m * N + n];
        if (flags & FLAG_SPLIT) {
          // (B*T, D) -> (B*H, T, DH)
          int b = m >> 12, t = m & (Tc - 1);
          int h = n >> 6,  dh = n & (DHc - 1);
          outF[(((size_t)(b * Hc + h)) * Tc + t) * DHc + dh] = v;
        } else {
          if (outF) outF[(size_t)m * N + n] = v;
          if (outB) outB[(size_t)m * N + n] = (__bf16)v;
        }
      }
    }
  }
}

// ---------------------------------------------------------------------------
// LSH bucket assignment: argmax over [R, -R] of qk . rot  per hash round
// ---------------------------------------------------------------------------
__global__ __launch_bounds__(256)
void bucket_kernel(const float* __restrict__ qk, const float* __restrict__ rot,
                   unsigned char* __restrict__ bkt)
{
  size_t idx = (size_t)blockIdx.x * 256 + threadIdx.x;   // bh*T = 131072
  int bh = (int)(idx >> 12), t = (int)(idx & (Tc - 1));
  float q[DHc];
  const float4* qp = (const float4*)(qk + (((size_t)bh) * Tc + t) * DHc);
  #pragma unroll
  for (int i = 0; i < 16; ++i) {
    float4 f = qp[i];
    q[4*i+0] = f.x; q[4*i+1] = f.y; q[4*i+2] = f.z; q[4*i+3] = f.w;
  }
  for (int r = 0; r < NHASHc; ++r) {
    float best = -3.0e38f; int bj = 0;
    for (int j = 0; j < NBc / 2; ++j) {
      float s = 0.f;
      #pragma unroll 8
      for (int f = 0; f < DHc; ++f) s += q[f] * rot[(f * NHASHc + r) * (NBc/2) + j];
      if ( s > best) { best =  s; bj = j; }
      if (-s > best) { best = -s; bj = j + NBc/2; }
    }
    bkt[(((size_t)bh) * NHASHc + r) * Tc + t] = (unsigned char)bj;
  }
}

// ---------------------------------------------------------------------------
// Stable counting sort into 64 buckets per (bh, round): st[i] = orig position
// ---------------------------------------------------------------------------
__global__ __launch_bounds__(256)
void sort_kernel(const unsigned char* __restrict__ bkt, int* __restrict__ st_all)
{
  __shared__ unsigned char lb[Tc];
  __shared__ int hist[NBc];
  __shared__ int offs[NBc];
  int tid = threadIdx.x;
  const unsigned char* src = bkt + (size_t)blockIdx.x * Tc;
  *(uint4*)(lb + tid * 16) = *(const uint4*)(src + tid * 16);   // 256*16 = 4096
  if (tid < NBc) hist[tid] = 0;
  __syncthreads();
  for (int i = tid; i < Tc; i += 256) atomicAdd(&hist[lb[i]], 1);
  __syncthreads();
  if (tid == 0) {
    int a = 0;
    for (int b = 0; b < NBc; ++b) { offs[b] = a; a += hist[b]; }
  }
  __syncthreads();
  if (tid < NBc) {
    int o = offs[tid];
    int* dst = st_all + (size_t)blockIdx.x * Tc;
    for (int t = 0; t < Tc; ++t)
      if (lb[t] == (unsigned char)tid) dst[o++] = t;
  }
}

// ---------------------------------------------------------------------------
// Chunked LSH attention with look-one-back; one workgroup per (chunk, bh).
// dots (64x128, K=64) and AV (64x64, K=128) via WMMA bf16.
// ---------------------------------------------------------------------------
__global__ __launch_bounds__(256)
void lsh_attn_kernel(const float* __restrict__ qk, const float* __restrict__ v,
                     const int* __restrict__ st_all,
                     float* __restrict__ o_all, float* __restrict__ lg_all)
{
  __shared__ __bf16 Qs[64 * 64];        // raw q rows,         ld 64
  __shared__ __bf16 Ks[128 * 64];       // normalized k rows,  ld 64
  __shared__ __bf16 VT[64 * 128];       // v transposed [n][k],ld 128
  __shared__ __bf16 DotsP[64 * 128];    // dots then probs,    ld 128
  __shared__ int pq[64];
  __shared__ int pk[128];

  int tid = threadIdx.x, lane = tid & 31, w = tid >> 5;
  int c  = blockIdx.x, bh = blockIdx.y;
  int r  = c >> 6,  cc  = c & (NBc - 1);
  int cp = (c + Cc - 1) & (Cc - 1);     // look-one-back over the global chunk axis
  int rp = cp >> 6, ccp = cp & (NBc - 1);

  if (tid < 64) {
    int p = st_all[((size_t)bh * NHASHc + r) * Tc + cc * BSc + tid];
    pq[tid] = p;  pk[tid] = p;
  } else if (tid < 128) {
    pk[tid] = st_all[((size_t)bh * NHASHc + rp) * Tc + ccp * BSc + (tid - 64)];
  }
  __syncthreads();

  // ---- gather q / k(normalized) / v^T into LDS (bf16) ----
  if (tid < 128) {
    int j = tid, pos = pk[j];
    const float* qrow = qk + (((size_t)bh) * Tc + pos) * DHc;
    float qr[DHc]; float ss = 0.f;
    #pragma unroll
    for (int f = 0; f < DHc; f += 4) {
      float4 t4 = *(const float4*)(qrow + f);
      qr[f] = t4.x; qr[f+1] = t4.y; qr[f+2] = t4.z; qr[f+3] = t4.w;
      ss += t4.x*t4.x + t4.y*t4.y + t4.z*t4.z + t4.w*t4.w;
    }
    float inv = 1.f / fmaxf(sqrtf(ss), 1e-12f);
    #pragma unroll
    for (int f = 0; f < DHc; ++f) Ks[j * 64 + f] = (__bf16)(qr[f] * inv);
    if (j < 64) {
      #pragma unroll
      for (int f = 0; f < DHc; ++f) Qs[j * 64 + f] = (__bf16)qr[f];
    }
    const float* vrow = v + (((size_t)bh) * Tc + pos) * DHc;
    #pragma unroll
    for (int f = 0; f < DHc; ++f) VT[f * 128 + j] = (__bf16)vrow[f];
  }
  __syncthreads();

  // ---- dots = q @ k^T : 64x128, K=64.  Wave w handles 16 cols j0..j0+15 ----
  {
    int j0 = w * 16;
    f32x8 acc[4];
    acc[0] = zero8(); acc[1] = zero8(); acc[2] = zero8(); acc[3] = zero8();
    #pragma unroll
    for (int k0 = 0; k0 < 64; k0 += 32) {
      bf16x16 fb = ldfrag(Ks + (size_t)j0 * 64 + k0, 64);
      #pragma unroll
      for (int mt = 0; mt < 4; ++mt) {
        bf16x16 fa = ldfrag(Qs + (size_t)(mt * 16) * 64 + k0, 64);
        acc[mt] = wmma_bf16(fa, fb, acc[mt]);
      }
    }
    int hi = lane >> 4, cl = lane & 15;
    #pragma unroll
    for (int mt = 0; mt < 4; ++mt) {
      #pragma unroll
      for (int i = 0; i < 8; ++i) {
        int row = mt * 16 + hi * 8 + i;
        int col = j0 + cl;
        float dv = acc[mt][i] * 0.125f;              // DH^-0.5
        if (pq[row] == pk[col]) dv = MASKV;          // self-attention mask
        DotsP[row * 128 + col] = (__bf16)dv;
      }
    }
  }
  __syncthreads();

  // ---- row softmax + logsumexp (threads 0..63 own one row each) ----
  if (tid < 64) {
    int i = tid;
    float m = -3.0e38f;
    for (int j = 0; j < 128; ++j) m = fmaxf(m, (float)DotsP[i * 128 + j]);
    float s = 0.f;
    for (int j = 0; j < 128; ++j) s += expf((float)DotsP[i * 128 + j] - m);
    float lse = m + logf(s);
    for (int j = 0; j < 128; ++j)
      DotsP[i * 128 + j] = (__bf16)expf((float)DotsP[i * 128 + j] - lse);
    lg_all[((size_t)bh * NHASHc + r) * Tc + pq[i]] = lse;
  }
  __syncthreads();

  // ---- bo = probs @ v : 64x64, K=128.  16 tiles / 8 waves ----
  int hi = lane >> 4, cl = lane & 15;
  #pragma unroll
  for (int tt = 0; tt < 2; ++tt) {
    int tile = w * 2 + tt, mt = tile >> 2, nt = tile & 3;
    f32x8 acc = zero8();
    #pragma unroll
    for (int k0 = 0; k0 < 128; k0 += 32) {
      bf16x16 fa = ldfrag(DotsP + (size_t)(mt * 16) * 128 + k0, 128);
      bf16x16 fb = ldfrag(VT    + (size_t)(nt * 16) * 128 + k0, 128);
      acc = wmma_bf16(fa, fb, acc);
    }
    #pragma unroll
    for (int i = 0; i < 8; ++i) {
      int row = mt * 16 + hi * 8 + i;
      int col = nt * 16 + cl;
      o_all[(((size_t)bh * NHASHc + r) * Tc + pq[row]) * DHc + col] = acc[i];
    }
  }
}

// ---------------------------------------------------------------------------
// Combine hash rounds: softmax over per-round logsumexp, merge heads.
// Emits bf16 directly (feeds only the Wout WMMA GEMM).
// ---------------------------------------------------------------------------
__global__ __launch_bounds__(256)
void combine_kernel(const float* __restrict__ o_all, const float* __restrict__ lg_all,
                    __bf16* __restrict__ outb)
{
  size_t idx = (size_t)blockIdx.x * 256 + threadIdx.x;  // bh*T
  int bh = (int)(idx >> 12), t = (int)(idx & (Tc - 1));
  float lg[NHASHc]; float m = -3.0e38f;
  #pragma unroll
  for (int r = 0; r < NHASHc; ++r) {
    lg[r] = lg_all[((size_t)bh * NHASHc + r) * Tc + t];
    m = fmaxf(m, lg[r]);
  }
  float s = 0.f;
  #pragma unroll
  for (int r = 0; r < NHASHc; ++r) { lg[r] = expf(lg[r] - m); s += lg[r]; }
  float inv = 1.f / s;
  int b = bh >> 3, h = bh & (Hc - 1);
  __bf16* op = outb + (((size_t)b * Tc) + t) * Dc + h * DHc;
  for (int dh = 0; dh < DHc; dh += 4) {
    float ax = 0.f, ay = 0.f, az = 0.f, aw = 0.f;
    #pragma unroll
    for (int r = 0; r < NHASHc; ++r) {
      const float4 ov = *(const float4*)(o_all +
          (((size_t)(bh * NHASHc + r)) * Tc + t) * DHc + dh);
      float wr = lg[r] * inv;
      ax += wr * ov.x; ay += wr * ov.y; az += wr * ov.z; aw += wr * ov.w;
    }
    op[dh + 0] = (__bf16)ax;
    op[dh + 1] = (__bf16)ay;
    op[dh + 2] = (__bf16)az;
    op[dh + 3] = (__bf16)aw;
  }
}

// ---------------------------------------------------------------------------
// LayerNorm over D=512 (one wave32 per row, 8 rows per block).
// Optionally dual-writes a bf16 shadow for the next layer's WMMA GEMMs.
// ---------------------------------------------------------------------------
__global__ __launch_bounds__(256)
void ln_kernel(const float* __restrict__ in, const float* __restrict__ g,
               const float* __restrict__ b, float* __restrict__ out,
               __bf16* __restrict__ outb)
{
  int tid = threadIdx.x, lane = tid & 31, w = tid >> 5;
  size_t row = (size_t)blockIdx.x * 8 + w;
  const float* x = in + row * Dc;
  float vals[16]; float s = 0.f, s2 = 0.f;
  #pragma unroll
  for (int i = 0; i < 16; ++i) {
    float vv = x[lane + i * 32];
    vals[i] = vv; s += vv; s2 += vv * vv;
  }
  #pragma unroll
  for (int off = 16; off; off >>= 1) {
    s  += __shfl_xor(s,  off, 32);
    s2 += __shfl_xor(s2, off, 32);
  }
  float mean = s * (1.f / Dc);
  float var  = s2 * (1.f / Dc) - mean * mean;
  float rstd = rsqrtf(var + 1e-6f);
  float* o = out + row * Dc;
  __bf16* ob = outb ? outb + row * Dc : nullptr;
  #pragma unroll
  for (int i = 0; i < 16; ++i) {
    int cidx = lane + i * 32;
    float vv = (vals[i] - mean) * rstd * g[cidx] + b[cidx];
    o[cidx] = vv;
    if (ob) ob[cidx] = (__bf16)vv;
  }
}

// ---------------------------------------------------------------------------
// Host orchestration
// ---------------------------------------------------------------------------
static void launch_cast(const float* in, __bf16* out, int n, hipStream_t s) {
  cast_f32_bf16_kernel<<<n / 1024, 256, 0, s>>>(in, out, n);
}

extern "C" void kernel_launch(void* const* d_in, const int* in_sizes, int n_in,
                              void* d_out, int out_size, void* d_ws, size_t ws_size,
                              hipStream_t stream) {
  (void)in_sizes; (void)n_in; (void)out_size; (void)ws_size;

  const float* src  = (const float*)d_in[0];
  const float* rots = (const float*)d_in[1];
  const float* Wqk  = (const float*)d_in[2];
  const float* Wv   = (const float*)d_in[3];
  const float* Wout = (const float*)d_in[4];
  const float* bout = (const float*)d_in[5];
  const float* W1   = (const float*)d_in[6];
  const float* b1   = (const float*)d_in[7];
  const float* W2   = (const float*)d_in[8];
  const float* b2   = (const float*)d_in[9];
  const float* g2   = (const float*)d_in[10];
  const float* be2  = (const float*)d_in[11];
  const float* gf   = (const float*)d_in[12];
  const float* bfin = (const float*)d_in[13];
  float* out = (float*)d_out;

  // ---- workspace carve ----
  char* p = (char*)d_ws;
  auto alloc = [&](size_t bytes) -> void* {
    void* r = (void*)p;
    p += (bytes + 255) & ~((size_t)255);
    return r;
  };
  const size_t XE = (size_t)Mrows * Dc;           // 8,388,608
  float* xbuf  = (float*)alloc(XE * 4);
  float* tmp   = (float*)alloc(XE * 4);
  float* qkbuf = (float*)alloc(XE * 4);
  float* vbuf  = (float*)alloc(XE * 4);
  float* o_all = (float*)alloc((size_t)BHc * NHASHc * Tc * DHc * 4);
  float* lg_all= (float*)alloc((size_t)BHc * NHASHc * Tc * 4);
  __bf16* xb   = (__bf16*)alloc(XE * 2);
  __bf16* attb = (__bf16*)alloc(XE * 2);
  __bf16* hb   = (__bf16*)alloc((size_t)Mrows * DFFc * 2);
  __bf16* Wqkb = (__bf16*)alloc((size_t)Lc * Dc * Dc * 2);
  __bf16* Wvb  = (__bf16*)alloc((size_t)Lc * Dc * Dc * 2);
  __bf16* Woutb= (__bf16*)alloc((size_t)Lc * Dc * Dc * 2);
  __bf16* W1b  = (__bf16*)alloc((size_t)Lc * DFFc * Dc * 2);
  __bf16* W2b  = (__bf16*)alloc((size_t)Lc * Dc * DFFc * 2);
  unsigned char* bkt = (unsigned char*)alloc((size_t)BHc * NHASHc * Tc);
  int* st_all  = (int*)alloc((size_t)BHc * NHASHc * Tc * 4);

  // ---- one-time per-call weight casts ----
  launch_cast(Wqk,  Wqkb,  Lc * Dc * Dc,   stream);
  launch_cast(Wv,   Wvb,   Lc * Dc * Dc,   stream);
  launch_cast(Wout, Woutb, Lc * Dc * Dc,   stream);
  launch_cast(W1,   W1b,   Lc * DFFc * Dc, stream);
  launch_cast(W2,   W2b,   Lc * Dc * DFFc, stream);

  hipMemcpyAsync(xbuf, src, XE * 4, hipMemcpyDeviceToDevice, stream);
  launch_cast(src, xb, (int)XE, stream);   // bf16 shadow of x, kept in sync below

  const dim3 blk(256);
  const dim3 gemmD(Dc / GBN, Mrows / GBM);     // N=512  -> (4,128)
  const dim3 gemmF(DFFc / GBN, Mrows / GBM);   // N=2048 -> (16,128)
  const dim3 attnG(Cc, BHc);

  for (int i = 0; i < Lc; ++i) {
    const size_t DD = (size_t)Dc * Dc, FD = (size_t)DFFc * Dc;

    // qk / v projections with head-split store
    gemm_bf16_kernel<<<gemmD, blk, 0, stream>>>(xb, Wqkb + i * DD, nullptr, nullptr,
                                                qkbuf, nullptr, Dc, Dc, FLAG_SPLIT);
    gemm_bf16_kernel<<<gemmD, blk, 0, stream>>>(xb, Wvb + i * DD, nullptr, nullptr,
                                                vbuf, nullptr, Dc, Dc, FLAG_SPLIT);
    // LSH buckets + stable sort per (bh, round)
    bucket_kernel<<<(BHc * Tc) / 256, blk, 0, stream>>>(
        qkbuf, rots + (size_t)i * DHc * NHASHc * (NBc / 2), bkt);
    sort_kernel<<<BHc * NHASHc, blk, 0, stream>>>(bkt, st_all);
    // chunked attention per (chunk, bh)
    lsh_attn_kernel<<<attnG, blk, 0, stream>>>(qkbuf, vbuf, st_all, o_all, lg_all);
    // combine rounds, merge heads -> bf16 directly
    combine_kernel<<<(BHc * Tc) / 256, blk, 0, stream>>>(o_all, lg_all, attb);
    // x = x + att @ Wout^T + bout   (dual-write f32 + bf16 shadow)
    gemm_bf16_kernel<<<gemmD, blk, 0, stream>>>(attb, Woutb + i * DD, bout + i * Dc,
                                                xbuf, xbuf, xb, Dc, Dc, 0);
    // FFN: h = relu(x@W1^T + b1) (bf16) ; tmp = x + h@W2^T + b2 (f32)
    gemm_bf16_kernel<<<gemmF, blk, 0, stream>>>(xb, W1b + i * FD, b1 + i * DFFc,
                                                nullptr, nullptr, hb, DFFc, Dc, FLAG_RELU);
    gemm_bf16_kernel<<<gemmD, blk, 0, stream>>>(hb, W2b + i * FD, b2 + i * Dc,
                                                xbuf, tmp, nullptr, Dc, DFFc, 0);
    // x = LN(tmp; g2, be2), dual-write bf16 shadow for next layer
    ln_kernel<<<Mrows / 8, blk, 0, stream>>>(tmp, g2 + i * Dc, be2 + i * Dc, xbuf, xb);
  }

  // final LayerNorm -> output
  ln_kernel<<<Mrows / 8, blk, 0, stream>>>(xbuf, gf, bfin, out, nullptr);
}
